// BondDistNet_38302518345941
// MI455X (gfx1250) — matrix operand order repaired and checked by reading, not compiled, not run on hardware
//
#include <hip/hip_runtime.h>

// ---------------------------------------------------------------------------
// Types for CDNA5 WMMA
// ---------------------------------------------------------------------------
typedef __attribute__((ext_vector_type(16))) __bf16 v16bf;
typedef __attribute__((ext_vector_type(8)))  __bf16 v8bf;
typedef __attribute__((ext_vector_type(8)))  float  v8f;

#define N_NODES 20000
#define N_GRAPH 1000
#define NPG     20
#define EBN     60000
#define EDN     380000
#define HID     512
#define OUTC    256
#define BN_EPS  1e-5f

// round-to-nearest-even f32 -> bf16 bits (avoids relying on __bf16 cast lowering)
__device__ __forceinline__ unsigned short f2bf_bits(float f) {
  unsigned int u = __float_as_uint(f);
  unsigned int lsb = (u >> 16) & 1u;
  u += 0x7fffu + lsb;
  return (unsigned short)(u >> 16);
}

// ---------------------------------------------------------------------------
// Elementwise / utility kernels
// ---------------------------------------------------------------------------
__global__ void fzero_kernel(float* __restrict__ p, long n) {
  long i = (long)blockIdx.x * blockDim.x + threadIdx.x;
  long stride = (long)gridDim.x * blockDim.x;
  for (; i < n; i += stride) p[i] = 0.f;
}

// src fp32 [R,C] (ld=C) -> dst bf16 [R,Cp] zero padded
__global__ void cvt_pad_kernel(const float* __restrict__ src, __bf16* __restrict__ dst,
                               int R, int C, int Cp) {
  size_t i = (size_t)blockIdx.x * blockDim.x + threadIdx.x;
  if (i >= (size_t)R * Cp) return;
  int c = (int)(i % Cp);
  int r = (int)(i / Cp);
  float v = (c < C) ? src[(size_t)r * C + c] : 0.f;
  ((unsigned short*)dst)[i] = f2bf_bits(v);
}

// src fp32 [R,C] -> dst bf16 [R,ldd] at column offset coff
__global__ void cvt_place_kernel(const float* __restrict__ src, __bf16* __restrict__ dst,
                                 int R, int C, int ldd, int coff) {
  size_t i = (size_t)blockIdx.x * blockDim.x + threadIdx.x;
  if (i >= (size_t)R * C) return;
  int c = (int)(i % C);
  int r = (int)(i / C);
  ((unsigned short*)dst)[(size_t)r * ldd + coff + c] = f2bf_bits(src[i]);
}

// xin[N,16] = [x(9) | pos_g(3) | pos_ex(3) | 0]
__global__ void build_xin_kernel(const float* __restrict__ x, const float* __restrict__ pg,
                                 const float* __restrict__ pe, float* __restrict__ xin) {
  int i = blockIdx.x * blockDim.x + threadIdx.x;
  if (i >= N_NODES * 16) return;
  int r = i >> 4, c = i & 15;
  float v = 0.f;
  if (c < 9)       v = x[r * 9 + c];
  else if (c < 12) v = pg[r * 3 + (c - 9)];
  else if (c < 15) v = pe[r * 3 + (c - 12)];
  xin[i] = v;
}

// dst bf16 [R,Cp] = bf16(agg[R,C] + x[R,ldx][:, :C]), zero pad
__global__ void add_cvt_pad_kernel(const float* __restrict__ agg, const float* __restrict__ x,
                                   int ldx, __bf16* __restrict__ dst, int R, int C, int Cp) {
  size_t i = (size_t)blockIdx.x * blockDim.x + threadIdx.x;
  if (i >= (size_t)R * Cp) return;
  int c = (int)(i % Cp);
  int r = (int)(i / Cp);
  float v = 0.f;
  if (c < C) v = agg[(size_t)r * C + c] + x[(size_t)r * ldx + c];
  ((unsigned short*)dst)[i] = f2bf_bits(v);
}

// ---------------------------------------------------------------------------
// GINE message + scatter-add (bandwidth bound; coalesced over channels)
// ---------------------------------------------------------------------------
__global__ void bond_msg_kernel(const float* __restrict__ x, int ldx, int C,
                                const int* __restrict__ ei, int E,
                                const float* __restrict__ ea,
                                const float* __restrict__ lw, const float* __restrict__ lb,
                                float* __restrict__ agg) {
  int cg = (C + 3) >> 2;
  long i = (long)blockIdx.x * blockDim.x + threadIdx.x;
  if (i >= (long)E * cg) return;
  int e  = (int)(i / cg);
  int c0 = (int)(i % cg) * 4;
  int s = ei[e], d = ei[E + e];
  float e0 = ea[e * 3 + 0], e1 = ea[e * 3 + 1], e2 = ea[e * 3 + 2];
#pragma unroll
  for (int j = 0; j < 4; ++j) {
    int c = c0 + j;
    if (c < C) {
      float m = x[(size_t)s * ldx + c] + e0 * lw[c * 3 + 0] + e1 * lw[c * 3 + 1]
              + e2 * lw[c * 3 + 2] + lb[c];
      m = m > 0.f ? m : 0.f;
      atomicAdd(&agg[(size_t)d * C + c], m);
    }
  }
}

__global__ void dist_msg_kernel(const float* __restrict__ x, int ldx, int C,
                                const int* __restrict__ ei, int E,
                                const float* __restrict__ eg, const float* __restrict__ ee,
                                const float* __restrict__ lw, const float* __restrict__ lb,
                                float* __restrict__ agg) {
  int cg = (C + 3) >> 2;
  long i = (long)blockIdx.x * blockDim.x + threadIdx.x;
  if (i >= (long)E * cg) return;
  int e  = (int)(i / cg);
  int c0 = (int)(i % cg) * 4;
  int s = ei[e], d = ei[E + e];
  float e0 = eg[e], e1 = ee[e];
#pragma unroll
  for (int j = 0; j < 4; ++j) {
    int c = c0 + j;
    if (c < C) {
      float m = x[(size_t)s * ldx + c] + e0 * lw[c * 2 + 0] + e1 * lw[c * 2 + 1] + lb[c];
      m = m > 0.f ? m : 0.f;
      atomicAdd(&agg[(size_t)d * C + c], m);
    }
  }
}

// ---------------------------------------------------------------------------
// BatchNorm column statistics (sum, sumsq). C must be a multiple of 64.
// ---------------------------------------------------------------------------
__global__ __launch_bounds__(256) void colstats_kernel(const float* __restrict__ h,
                                                       int R, int C,
                                                       float* __restrict__ sum,
                                                       float* __restrict__ sumsq) {
  const int cx = threadIdx.x & 63;
  const int ry = threadIdx.x >> 6;
  const int c  = blockIdx.x * 64 + cx;
  int r0 = blockIdx.y * 1024;
  int r1 = r0 + 1024; if (r1 > R) r1 = R;
  float s = 0.f, q = 0.f;
  for (int r = r0 + ry; r < r1; r += 4) {
    float v = h[(size_t)r * C + c];
    s += v; q += v * v;
  }
  __shared__ float ss[256], sq[256];
  ss[threadIdx.x] = s; sq[threadIdx.x] = q;
  __syncthreads();
  if (ry == 0) {
    s = ss[cx] + ss[64 + cx] + ss[128 + cx] + ss[192 + cx];
    q = sq[cx] + sq[64 + cx] + sq[128 + cx] + sq[192 + cx];
    atomicAdd(&sum[c], s);
    atomicAdd(&sumsq[c], q);
  }
}

// y = bf16(relu((h - mean) * rsqrt(var + eps) * g + be)); bias cancels under BN
__global__ void bn_relu_cvt_kernel(const float* __restrict__ h,
                                   const float* __restrict__ sum,
                                   const float* __restrict__ sumsq,
                                   const float* __restrict__ g, const float* __restrict__ be,
                                   float invR, __bf16* __restrict__ dst,
                                   int R, int C, int Cp) {
  size_t i = (size_t)blockIdx.x * blockDim.x + threadIdx.x;
  if (i >= (size_t)R * Cp) return;
  int c = (int)(i % Cp);
  int r = (int)(i / Cp);
  float y = 0.f;
  if (c < C) {
    float mean = sum[c] * invR;
    float var  = sumsq[c] * invR - mean * mean;
    float xv   = h[(size_t)r * C + c];
    y = (xv - mean) * rsqrtf(var + BN_EPS) * g[c] + be[c];
    y = y > 0.f ? y : 0.f;
  }
  ((unsigned short*)dst)[i] = f2bf_bits(y);
}

// graph mean pool (20 contiguous nodes per graph): pooled [G, 768] = [x1|x2|x3]
__global__ void pool_kernel(const float* __restrict__ x1, const float* __restrict__ x2,
                            const float* __restrict__ x3, float* __restrict__ pooled) {
  int i = blockIdx.x * blockDim.x + threadIdx.x;
  if (i >= N_GRAPH * 768) return;
  int gph = i / 768, c = i % 768;
  const float* p = (c < 256) ? x1 : (c < 512) ? x2 : x3;
  int cc = c & 255;
  float s = 0.f;
#pragma unroll
  for (int r = 0; r < NPG; ++r) s += p[(size_t)(gph * NPG + r) * 256 + cc];
  pooled[i] = s * (1.f / NPG);
}

// ---------------------------------------------------------------------------
// bf16 WMMA GEMM:  C[M,Nc] = A[M,K] @ B[Nc,K]^T  (+bias)(+relu)
// Block: 256 threads = 8 waves; 128x128 block tile; 32x64 per-wave tile
// (8 WMMAs / K-step / wave); double-buffered LDS, one barrier per K-step.
// EPI: 0 = raw store, 1 = +bias, 2 = +bias then relu
// ---------------------------------------------------------------------------
template<int EPI>
__global__ __launch_bounds__(256) void gemm_wmma_kernel(
    const __bf16* __restrict__ A, int lda,
    const __bf16* __restrict__ B, int ldb,
    const float* __restrict__ bias,
    float* __restrict__ C, int ldc,
    int M, int Nc, int K) {
  __shared__ __align__(64) __bf16 As[2][128][32];
  __shared__ __align__(64) __bf16 Bs[2][128][32];
  const int tid  = threadIdx.x;
  const int lane = tid & 31;
  const int wave = tid >> 5;
  const int wr = wave & 3;   // 4 row groups of 32
  const int wc = wave >> 2;  // 2 col groups of 64
  const int m0 = blockIdx.x * 128;
  const int n0 = blockIdx.y * 128;

  v8f acc[2][4];
#pragma unroll
  for (int s = 0; s < 2; ++s)
#pragma unroll
    for (int t = 0; t < 4; ++t)
      acc[s][t] = (v8f){0.f, 0.f, 0.f, 0.f, 0.f, 0.f, 0.f, 0.f};

  // cooperative loader coords: thread covers (row rl, cols cl..cl+7) in both halves
  const int rl = tid >> 2;        // 0..63
  const int cl = (tid & 3) * 8;   // 0,8,16,24

  uint4 avreg[2], bvreg[2];

  auto fetch = [&](int k0) {
#pragma unroll
    for (int p = 0; p < 2; ++p) {
      int r = rl + p * 64;
      uint4 av; av.x = av.y = av.z = av.w = 0u;
      if (m0 + r < M) av = *(const uint4*)(A + (size_t)(m0 + r) * lda + (k0 + cl));
      avreg[p] = av;
      uint4 bv; bv.x = bv.y = bv.z = bv.w = 0u;
      if (n0 + r < Nc) bv = *(const uint4*)(B + (size_t)(n0 + r) * ldb + (k0 + cl));
      bvreg[p] = bv;
    }
  };
  auto stage = [&](int buf) {
#pragma unroll
    for (int p = 0; p < 2; ++p) {
      *(uint4*)&As[buf][rl + p * 64][cl] = avreg[p];
      *(uint4*)&Bs[buf][rl + p * 64][cl] = bvreg[p];
    }
  };

  fetch(0);
  stage(0);
  __syncthreads();

  const int nsteps = K >> 5;  // K multiple of 32
  for (int kk = 0; kk < nsteps; ++kk) {
    const int cur = kk & 1;
    if (kk + 1 < nsteps) fetch((kk + 1) << 5);
    if (kk + 2 < nsteps && m0 + rl < M)
      __builtin_prefetch(A + (size_t)(m0 + rl) * lda + (((kk + 2) << 5) + cl), 0, 3);

    // A fragments (16x32 each): lane<16 row=lane K=0..15; lane>=16 K=16..31
    const int kba = (lane < 16) ? 0 : 16;
    const int ml  = wr * 32 + (lane & 15);
    v16bf af[2];
#pragma unroll
    for (int s = 0; s < 2; ++s) {
      v8bf lo = *(const v8bf*)&As[cur][ml + s * 16][kba];
      v8bf hi = *(const v8bf*)&As[cur][ml + s * 16][kba + 8];
#pragma unroll
      for (int i = 0; i < 8; ++i) { af[s][i] = lo[i]; af[s][i + 8] = hi[i]; }
    }
    // B fragments (32x16): lane<16 -> K=0..15, lane>=16 -> K=16..31; N = lane&15
    const int kbb = (lane < 16) ? 0 : 16;
    const int nl  = wc * 64 + (lane & 15);
    v16bf bq[4];
#pragma unroll
    for (int t = 0; t < 4; ++t) bq[t] = *(const v16bf*)&Bs[cur][nl + t * 16][kbb];

#pragma unroll
    for (int s = 0; s < 2; ++s)
#pragma unroll
      for (int t = 0; t < 4; ++t)
        acc[s][t] = __builtin_amdgcn_wmma_f32_16x16x32_bf16(false, af[s], false, bq[t],
                                                            (short)0, acc[s][t], false, false);

    if (kk + 1 < nsteps) stage(cur ^ 1);
    __syncthreads();
  }

  // C/D layout: VGPR r -> M = r + (lane>=16 ? 8 : 0), N = lane&15
  const int mb = m0 + wr * 32 + ((lane < 16) ? 0 : 8);
  const int nb = n0 + wc * 64 + (lane & 15);
#pragma unroll
  for (int t = 0; t < 4; ++t) {
    int n = nb + t * 16;
    if (n >= Nc) continue;
    float bv = (EPI >= 1) ? bias[n] : 0.f;
#pragma unroll
    for (int s = 0; s < 2; ++s) {
#pragma unroll
      for (int r = 0; r < 8; ++r) {
        int m = mb + s * 16 + r;
        if (m >= M) continue;
        float v = acc[s][t][r] + bv;
        if (EPI == 2) v = v > 0.f ? v : 0.f;
        C[(size_t)m * ldc + n] = v;
      }
    }
  }
}

// ---------------------------------------------------------------------------
// Host orchestration
// ---------------------------------------------------------------------------
struct LayerP {
  const float *bond_lb, *bond_lw, *b_b1, *b_b2, *b_be, *b_g, *b_w1, *b_w2;
  const float *dist_lb, *dist_lw, *d_b1, *d_b2, *d_be, *d_g, *d_w1, *d_w2;
  const float *fc_b, *fc_w;
};

extern "C" void kernel_launch(void* const* d_in, const int* in_sizes, int n_in,
                              void* d_out, int out_size, void* d_ws, size_t ws_size,
                              hipStream_t stream) {
  (void)in_sizes; (void)n_in; (void)out_size; (void)ws_size;

  const float* x      = (const float*)d_in[0];
  const float* pos_g  = (const float*)d_in[1];
  const float* pos_ex = (const float*)d_in[2];
  const int*   ei     = (const int*)d_in[3];   // [2, EB]
  const float* ea     = (const float*)d_in[4]; // [EB, 3]
  const int*   fei    = (const int*)d_in[5];   // [2, ED]
  const float* deg    = (const float*)d_in[6]; // [ED]
  const float* dee    = (const float*)d_in[7]; // [ED]
  // d_in[8] = batch (contiguous 20-node graphs; exploited structurally)

  // params flattened as a JAX pytree (dicts sorted by key):
  const float* fc1_b  = (const float*)d_in[9];
  const float* fc1_be = (const float*)d_in[10];
  const float* fc1_g  = (const float*)d_in[11];
  const float* fc1_w  = (const float*)d_in[12]; // [256, 768]
  const float* fc2_b  = (const float*)d_in[13];
  const float* fc2_be = (const float*)d_in[14];
  const float* fc2_g  = (const float*)d_in[15];
  const float* fc2_w  = (const float*)d_in[16]; // [256, 256]
  const float* fcl_b  = (const float*)d_in[17]; // [2]
  const float* fcl_w  = (const float*)d_in[18]; // [2, 256]
  (void)fc1_b; (void)fc2_b; // cancel under training-mode BN

  LayerP LP[3];
  for (int l = 0; l < 3; ++l) {
    int b = 19 + 18 * l;
    LP[l].bond_lb = (const float*)d_in[b + 0];
    LP[l].bond_lw = (const float*)d_in[b + 1];
    LP[l].b_b1    = (const float*)d_in[b + 2];
    LP[l].b_b2    = (const float*)d_in[b + 3];
    LP[l].b_be    = (const float*)d_in[b + 4];
    LP[l].b_g     = (const float*)d_in[b + 5];
    LP[l].b_w1    = (const float*)d_in[b + 6];
    LP[l].b_w2    = (const float*)d_in[b + 7];
    LP[l].dist_lb = (const float*)d_in[b + 8];
    LP[l].dist_lw = (const float*)d_in[b + 9];
    LP[l].d_b1    = (const float*)d_in[b + 10];
    LP[l].d_b2    = (const float*)d_in[b + 11];
    LP[l].d_be    = (const float*)d_in[b + 12];
    LP[l].d_g     = (const float*)d_in[b + 13];
    LP[l].d_w1    = (const float*)d_in[b + 14];
    LP[l].d_w2    = (const float*)d_in[b + 15];
    LP[l].fc_b    = (const float*)d_in[b + 16];
    LP[l].fc_w    = (const float*)d_in[b + 17];
  }

  // ---- workspace bump allocator ----
  char* base = (char*)d_ws;
  size_t off = 0;
  auto alloc = [&](size_t bytes) -> void* {
    void* p = base + off;
    off = (off + bytes + 255) & ~(size_t)255;
    return p;
  };

  float* xin   = (float*)alloc((size_t)N_NODES * 16 * 4);
  float* x1    = (float*)alloc((size_t)N_NODES * OUTC * 4);
  float* x2    = (float*)alloc((size_t)N_NODES * OUTC * 4);
  float* x3    = (float*)alloc((size_t)N_NODES * OUTC * 4);
  float* agg   = (float*)alloc((size_t)N_NODES * OUTC * 4);
  __bf16* u_bf = (__bf16*)alloc((size_t)N_NODES * OUTC * 2);
  float* h1    = (float*)alloc((size_t)N_NODES * HID * 4);
  __bf16* h1bf = (__bf16*)alloc((size_t)N_NODES * HID * 2);
  float* bx    = (float*)alloc((size_t)N_NODES * OUTC * 4);
  float* dxm   = (float*)alloc((size_t)N_NODES * OUTC * 4);
  __bf16* catb = (__bf16*)alloc((size_t)N_NODES * 512 * 2);
  float* stats = (float*)alloc(1024 * 4);   // sum[512] | sumsq[512]
  float* sum   = stats;
  float* sq    = stats + 512;
  float* pooled   = (float*)alloc((size_t)N_GRAPH * 768 * 4);
  __bf16* pooledb = (__bf16*)alloc((size_t)N_GRAPH * 768 * 2);
  float* hg       = (float*)alloc((size_t)N_GRAPH * 256 * 4);
  __bf16* hgbf    = (__bf16*)alloc((size_t)N_GRAPH * 256 * 2);
  __bf16* w1b_bf  = (__bf16*)alloc((size_t)HID * 256 * 2);
  __bf16* w1d_bf  = (__bf16*)alloc((size_t)HID * 256 * 2);
  __bf16* w2b_bf  = (__bf16*)alloc((size_t)OUTC * HID * 2);
  __bf16* w2d_bf  = (__bf16*)alloc((size_t)OUTC * HID * 2);
  __bf16* fcw_bf  = (__bf16*)alloc((size_t)OUTC * 512 * 2);
  __bf16* fc1w_bf = (__bf16*)alloc((size_t)256 * 768 * 2);
  __bf16* fc2w_bf = (__bf16*)alloc((size_t)256 * 256 * 2);
  __bf16* fclw_bf = (__bf16*)alloc((size_t)2 * 256 * 2);

  auto zero = [&](float* p, long n) {
    long blocks = (n + 255) / 256;
    if (blocks > 4096) blocks = 4096;
    fzero_kernel<<<dim3((unsigned)blocks), 256, 0, stream>>>(p, n);
  };
  auto cvt_pad = [&](const float* s, __bf16* d, int R, int C, int Cp) {
    long tot = (long)R * Cp;
    cvt_pad_kernel<<<dim3((unsigned)((tot + 255) / 256)), 256, 0, stream>>>(s, d, R, C, Cp);
  };
  auto gemm = [&](const __bf16* A, int lda, const __bf16* B, int ldb, const float* bias,
                  float* Cm, int ldc, int M, int Nc, int K, int epi) {
    dim3 grid((M + 127) / 128, (Nc + 127) / 128);
    if (epi == 0)
      gemm_wmma_kernel<0><<<grid, 256, 0, stream>>>(A, lda, B, ldb, bias, Cm, ldc, M, Nc, K);
    else if (epi == 1)
      gemm_wmma_kernel<1><<<grid, 256, 0, stream>>>(A, lda, B, ldb, bias, Cm, ldc, M, Nc, K);
    else
      gemm_wmma_kernel<2><<<grid, 256, 0, stream>>>(A, lda, B, ldb, bias, Cm, ldc, M, Nc, K);
  };

  const float invN = 1.f / (float)N_NODES;
  const float invG = 1.f / (float)N_GRAPH;

  build_xin_kernel<<<dim3((N_NODES * 16 + 255) / 256), 256, 0, stream>>>(x, pos_g, pos_ex, xin);

  const int kpads[3] = {32, 256, 256};
  float* xouts[3] = {x1, x2, x3};

  for (int l = 0; l < 3; ++l) {
    const LayerP& P = LP[l];
    const float* xb; int lxb, Cb;
    const float* xd; int lxd, Cd;
    if (l == 0) { xb = xin; lxb = 16; Cb = 15; xd = x; lxd = 9; Cd = 9; }
    else        { xb = xd = xouts[l - 1]; lxb = lxd = 256; Cb = Cd = 256; }
    const int Kp = kpads[l];

    // weights -> bf16 (tiny; done per call, deterministic)
    cvt_pad(P.b_w1, w1b_bf, HID, Cb, Kp);
    cvt_pad(P.d_w1, w1d_bf, HID, Cd, Kp);
    cvt_pad(P.b_w2, w2b_bf, OUTC, HID, HID);
    cvt_pad(P.d_w2, w2d_bf, OUTC, HID, HID);
    cvt_pad(P.fc_w, fcw_bf, OUTC, 512, 512);

    // ---- bond branch -> bx ----
    zero(agg, (long)N_NODES * Cb);
    {
      long tot = (long)EBN * ((Cb + 3) / 4);
      bond_msg_kernel<<<dim3((unsigned)((tot + 255) / 256)), 256, 0, stream>>>(
          xb, lxb, Cb, ei, EBN, ea, P.bond_lw, P.bond_lb, agg);
    }
    add_cvt_pad_kernel<<<dim3((unsigned)(((long)N_NODES * Kp + 255) / 256)), 256, 0, stream>>>(
        agg, xb, lxb, u_bf, N_NODES, Cb, Kp);
    gemm(u_bf, Kp, w1b_bf, Kp, nullptr, h1, HID, N_NODES, HID, Kp, 0);
    zero(stats, 1024);
    colstats_kernel<<<dim3(HID / 64, (N_NODES + 1023) / 1024), 256, 0, stream>>>(h1, N_NODES, HID, sum, sq);
    bn_relu_cvt_kernel<<<dim3((unsigned)(((long)N_NODES * HID + 255) / 256)), 256, 0, stream>>>(
        h1, sum, sq, P.b_g, P.b_be, invN, h1bf, N_NODES, HID, HID);
    gemm(h1bf, HID, w2b_bf, HID, P.b_b2, bx, OUTC, N_NODES, OUTC, HID, 2);

    // ---- dist branch -> dxm ----
    zero(agg, (long)N_NODES * Cd);
    {
      long tot = (long)EDN * ((Cd + 3) / 4);
      dist_msg_kernel<<<dim3((unsigned)((tot + 255) / 256)), 256, 0, stream>>>(
          xd, lxd, Cd, fei, EDN, deg, dee, P.dist_lw, P.dist_lb, agg);
    }
    add_cvt_pad_kernel<<<dim3((unsigned)(((long)N_NODES * Kp + 255) / 256)), 256, 0, stream>>>(
        agg, xd, lxd, u_bf, N_NODES, Cd, Kp);
    gemm(u_bf, Kp, w1d_bf, Kp, nullptr, h1, HID, N_NODES, HID, Kp, 0);
    zero(stats, 1024);
    colstats_kernel<<<dim3(HID / 64, (N_NODES + 1023) / 1024), 256, 0, stream>>>(h1, N_NODES, HID, sum, sq);
    bn_relu_cvt_kernel<<<dim3((unsigned)(((long)N_NODES * HID + 255) / 256)), 256, 0, stream>>>(
        h1, sum, sq, P.d_g, P.d_be, invN, h1bf, N_NODES, HID, HID);
    gemm(h1bf, HID, w2d_bf, HID, P.d_b2, dxm, OUTC, N_NODES, OUTC, HID, 2);

    // ---- combine: [bx|dx] @ fc_w^T + fc_b -> xout ----
    cvt_place_kernel<<<dim3((unsigned)(((long)N_NODES * 256 + 255) / 256)), 256, 0, stream>>>(
        bx, catb, N_NODES, 256, 512, 0);
    cvt_place_kernel<<<dim3((unsigned)(((long)N_NODES * 256 + 255) / 256)), 256, 0, stream>>>(
        dxm, catb, N_NODES, 256, 512, 256);
    gemm(catb, 512, fcw_bf, 512, P.fc_b, xouts[l], OUTC, N_NODES, OUTC, 512, 1);
  }

  // ---- readout ----
  pool_kernel<<<dim3((N_GRAPH * 768 + 255) / 256), 256, 0, stream>>>(x1, x2, x3, pooled);
  cvt_pad(pooled, pooledb, N_GRAPH, 768, 768);

  cvt_pad(fc1_w, fc1w_bf, 256, 768, 768);
  gemm(pooledb, 768, fc1w_bf, 768, nullptr, hg, 256, N_GRAPH, 256, 768, 0);
  zero(stats, 1024);
  colstats_kernel<<<dim3(256 / 64, 1), 256, 0, stream>>>(hg, N_GRAPH, 256, sum, sq);
  bn_relu_cvt_kernel<<<dim3((unsigned)(((long)N_GRAPH * 256 + 255) / 256)), 256, 0, stream>>>(
      hg, sum, sq, fc1_g, fc1_be, invG, hgbf, N_GRAPH, 256, 256);

  cvt_pad(fc2_w, fc2w_bf, 256, 256, 256);
  gemm(hgbf, 256, fc2w_bf, 256, nullptr, hg, 256, N_GRAPH, 256, 256, 0);
  zero(stats, 1024);
  colstats_kernel<<<dim3(256 / 64, 1), 256, 0, stream>>>(hg, N_GRAPH, 256, sum, sq);
  bn_relu_cvt_kernel<<<dim3((unsigned)(((long)N_GRAPH * 256 + 255) / 256)), 256, 0, stream>>>(
      hg, sum, sq, fc2_g, fc2_be, invG, hgbf, N_GRAPH, 256, 256);

  cvt_pad(fcl_w, fclw_bf, 2, 256, 256);
  gemm(hgbf, 256, fclw_bf, 256, fcl_b, (float*)d_out, 2, N_GRAPH, 2, 256, 1);
}
